// KPConv_7112465842598
// MI455X (gfx1250) — matrix operand order repaired and checked by reading, not compiled, not run on hardware
//
#include <hip/hip_runtime.h>
#include <stdint.h>

// Problem constants (match reference)
#define B_      2
#define S_      8192
#define CIN     64
#define COUT    64
#define KP      15
#define KP16    16     // padded kernel-point count for 16-row WMMA tiles
#define NS      32
#define RADIUS_ 1.0f
#define R2_     1.0f

typedef __attribute__((ext_vector_type(16))) _Float16 v16h;
typedef __attribute__((ext_vector_type(8)))  _Float16 v8h;
typedef __attribute__((ext_vector_type(8)))  float    v8f;

__device__ __forceinline__ v16h cat8(v8h lo, v8h hi) {
  return __builtin_shufflevector(lo, hi, 0, 1, 2, 3, 4, 5, 6, 7,
                                 8, 9, 10, 11, 12, 13, 14, 15);
}

// ---------------------------------------------------------------------------
// Kernel 0: weights (15,64,64) f32 -> f16, pre-packed into WMMA B-fragment
// order for stage-2:  whPacked[ch][otile][lane][e]  (32*4*32*16 = 64K halves).
// B-layout (this code's contiguous k order, 32x16): lane col N = L&15; lanes<16 hold K=0..15,
// lanes>=16 hold K=16..31; element e -> K = kb + e.
// Row k=15 of the padded K16 dimension is zero.
// ---------------------------------------------------------------------------
__global__ void kp_prep_w(const float* __restrict__ w,
                          _Float16* __restrict__ whPacked) {
  int i = blockIdx.x * blockDim.x + threadIdx.x;   // 0..65535
  const int e  = i & 15;
  const int L  = (i >> 4) & 31;
  const int ot = (i >> 9) & 3;
  const int ch = i >> 11;                          // 0..31
  const int kc = ch * 32 + ((L >= 16) ? 16 : 0) + e;   // 0..1023
  const int o  = ot * 16 + (L & 15);
  const int widx = kc * (CIN) + o;                 // flat (k*64+c)*64+o
  float v = (kc < KP * CIN) ? w[widx] : 0.0f;      // pad k=15 rows with 0
  whPacked[i] = (_Float16)v;
}

// ---------------------------------------------------------------------------
// Kernel 1: fused ball-query top-32 selection + influence computation.
// One query per lane; point-cloud xyz streamed through LDS in 2048-pt tiles
// (whole cloud is L2-resident, 96 KB/batch). Top-32 via max-replace buffer in
// LDS (order-invariant vs reference argsort: aggregation sums over neighbors).
// Out-of-radius slots are replaced by the running argmin index (== sorted[0]).
// Emits f16 influence in WMMA-A layout infl[q][k][n] (16B-aligned 8-runs).
// ---------------------------------------------------------------------------
#define QT   64      // queries per block (threads)
#define TILE 2048    // candidate points per LDS tile
#define SELP 33      // padded stride to dodge LDS bank conflicts

__global__ void __launch_bounds__(QT) kp_select(const float* __restrict__ xyz,
                                                const float* __restrict__ kpts,
                                                int* __restrict__ idxBuf,
                                                _Float16* __restrict__ inflBuf) {
  __shared__ float px[TILE], py[TILE], pz[TILE];   // 24 KB
  __shared__ float selD[QT * SELP];                // 8.25 KB
  __shared__ int   selI[QT * SELP];                // 8.25 KB
  __shared__ float kx[KP], ky[KP], kz[KP];

  const int tid = threadIdx.x;
  const int b   = blockIdx.x / (S_ / QT);
  const int qs  = (blockIdx.x % (S_ / QT)) * QT + tid;   // query index in batch
  const int qg  = b * S_ + qs;                            // global query index
  const float* bx = xyz + (size_t)b * S_ * 3;

  if (tid < KP) {
    kx[tid] = kpts[tid * 3 + 0];
    ky[tid] = kpts[tid * 3 + 1];
    kz[tid] = kpts[tid * 3 + 2];
  }
  for (int s = 0; s < NS; ++s) {
    selD[tid * SELP + s] = 1e30f;
    selI[tid * SELP + s] = 0;
  }

  const float qx = bx[qs * 3 + 0];
  const float qy = bx[qs * 3 + 1];
  const float qz = bx[qs * 3 + 2];

  float curMax = 1e30f; int curSlot = 0;     // max of current top-32
  float bestD  = 1e30f; int bestI   = 0;     // running argmin (== self)

  for (int t = 0; t < S_; t += TILE) {
    __syncthreads();
    for (int i = tid; i < TILE; i += QT) {
      const float* p = bx + (size_t)(t + i) * 3;
      px[i] = p[0]; py[i] = p[1]; pz[i] = p[2];
    }
    __syncthreads();
    #pragma unroll 4
    for (int j = 0; j < TILE; ++j) {
      float dx = qx - px[j], dy = qy - py[j], dz = qz - pz[j];
      float d  = fmaf(dx, dx, fmaf(dy, dy, dz * dz));
      int   jj = t + j;
      if (d < bestD) { bestD = d; bestI = jj; }
      if (d < curMax) {
        selD[tid * SELP + curSlot] = d;
        selI[tid * SELP + curSlot] = jj;
        float mx = -1.0f; int sl = 0;        // rescan for new max (rare path)
        for (int s = 0; s < NS; ++s) {
          float v = selD[tid * SELP + s];
          if (v > mx) { mx = v; sl = s; }
        }
        curMax = mx; curSlot = sl;
      }
    }
  }

  const float invR = 1.0f / (RADIUS_ + 1e-8f);
  for (int s = 0; s < NS; ++s) {
    float ds = selD[tid * SELP + s];
    int   ii = selI[tid * SELP + s];
    if (ds > R2_) ii = bestI;                // radius mask -> nearest (self)
    idxBuf[(size_t)qg * NS + s] = ii;
    const float* np = bx + (size_t)ii * 3;
    float rx = np[0] - qx, ry = np[1] - qy, rz = np[2] - qz;
    #pragma unroll
    for (int k = 0; k < KP16; ++k) {
      _Float16 h = (_Float16)0.0f;
      if (k < KP) {
        float ddx = rx - kx[k], ddy = ry - ky[k], ddz = rz - kz[k];
        float sq  = fmaxf(ddx * ddx + ddy * ddy + ddz * ddz, 0.0f);
        float inf = fmaxf(1.0f - sqrtf(sq + 1e-8f) * invR, 0.0f);
        h = (_Float16)inf;
      }
      inflBuf[(size_t)qg * (KP16 * NS) + k * NS + s] = h;  // [q][k][n]
    }
  }
}

// ---------------------------------------------------------------------------
// Kernel 2: WMMA double-GEMM. 16 queries / block, 4 wave32 waves.
// Stage 1 (per query):  agg(16k x 64c) = infl^T(16x32) x F(32x64)   [4 wmma]
// Stage 2 (per block):  out(16q x 64o) = agg(16x1024) x W(1024x64)  [128 wmma]
// 16-bit operand layouts per CDNA5 ISA 7.12.2:
//   A (16x32 f16): lane m = L&15; lanes<16 hold K {0-7,16-23}, lanes>=16 hold
//                  K {8-15,24-31} as v16h elements [0-7],[8-15].
//   B (32x16 f16): lane col N = L&15; A and B here: lanes<16 K=0-15, lanes>=16 K=16-31 (not ISA order).
//   C/D (f32)    : VGPR r -> row M = r (lanes<16) / r+8 (lanes>=16), N = L&15.
// Feature slice is kept transposed in LDS ([c][n]) so B fragments are
// contiguous 32B ds_load_b128 pairs; stage-2 B comes pre-packed from global.
// ---------------------------------------------------------------------------
__global__ void __launch_bounds__(128) kp_gemm(const float* __restrict__ feat,
                                               const int* __restrict__ idxBuf,
                                               const _Float16* __restrict__ inflBuf,
                                               const _Float16* __restrict__ whPacked,
                                               float* __restrict__ outp) {
  __shared__ _Float16 featT[4][CIN][NS];       // 16 KB : per-wave F^T slice
  __shared__ _Float16 aggLDS[16 * 1024];       // 32 KB : agg[qLocal][k*64+c]

  const int tid   = threadIdx.x;
  const int wave  = tid >> 5;
  const int lane  = tid & 31;
  const int l16   = lane & 15;
  const int hi    = (lane >= 16);              // lane-half selector
  const int nb0   = hi ? 8 : 0;                // A-fragment K sub-base
  const int kb    = hi ? 16 : 0;               // B-fragment K base
  const int qBase = blockIdx.x * 16;

  // ---------------- Stage 1: per-query agg = infl^T x F -------------------
  for (int iter = 0; iter < 4; ++iter) {
    const int q = qBase + wave * 4 + iter;     // this wave's query
    // Gather the 32 neighbors' features (f32->f16), transposed to [c][n]:
    // all 32 lanes write consecutive n addresses per channel (conflict-free).
    {
      const int n  = lane;
      const int nb = idxBuf[(size_t)q * NS + n];
      const float4* fp =
          (const float4*)(feat + ((size_t)(q / S_) * S_ + nb) * CIN);
      #pragma unroll
      for (int c4 = 0; c4 < CIN / 4; ++c4) {
        float4 v = fp[c4];
        featT[wave][c4 * 4 + 0][n] = (_Float16)v.x;
        featT[wave][c4 * 4 + 1][n] = (_Float16)v.y;
        featT[wave][c4 * 4 + 2][n] = (_Float16)v.z;
        featT[wave][c4 * 4 + 3][n] = (_Float16)v.w;
      }
    }
    __syncthreads();

    // A = infl^T : A[m=k][K=n] = inflBuf[q][m][K]  (two 16B global loads)
    v16h a;
    {
      const _Float16* ip = inflBuf + (size_t)q * (KP16 * NS) + l16 * NS;
      v8h alo = *(const v8h*)(ip + kb);
      v8h ahi = *(const v8h*)(ip + kb + 8);
      a = cat8(alo, ahi);
    }
    #pragma unroll
    for (int c0 = 0; c0 < CIN; c0 += 16) {
      // B fragment: featT[col][kb..kb+15] contiguous (2x ds_load_b128)
      v16h bm = *(const v16h*)&featT[wave][c0 + l16][kb];
      v8f c = {};
      c = __builtin_amdgcn_wmma_f32_16x16x32_f16(false, a, false, bm,
                                                 (short)0, c, false, false);
      const int mbase = hi ? 8 : 0;
      #pragma unroll
      for (int r = 0; r < 8; ++r)
        aggLDS[(q - qBase) * 1024 + (mbase + r) * CIN + c0 + l16] =
            (_Float16)c[r];
    }
    __syncthreads();
  }

  __syncthreads();   // aggLDS fully written by all waves

  // ---------------- Stage 2: out = aggflat x Wflat (barrier-free loop) -----
  const int oBase = wave * 16;                 // each wave owns one N-tile
  const _Float16* apBase = aggLDS + (size_t)l16 * 1024;
  v8f acc = {};
  #pragma unroll 8
  for (int ch = 0; ch < 32; ++ch) {            // K = 1024 in chunks of 32
    const int kc0 = ch * NS;
    v8h alo = *(const v8h*)(apBase + kc0 + kb);
    v8h ahi = *(const v8h*)(apBase + kc0 + kb + 8);
    v16h a  = cat8(alo, ahi);
    // Pre-packed B fragment: one contiguous 32B global load per lane,
    // L2/WGP$-resident (all blocks share the same 128 KB of weights).
    v16h bm = *(const v16h*)(whPacked + ((size_t)(ch * 4 + wave) * 32 + lane) * 16);
    acc = __builtin_amdgcn_wmma_f32_16x16x32_f16(false, a, false, bm,
                                                 (short)0, acc, false, false);
  }

  // Write the 16x16 output tile (f32).
  {
    const int mbase = hi ? 8 : 0;
    #pragma unroll
    for (int r = 0; r < 8; ++r) {
      const int qg = qBase + mbase + r;
      outp[(size_t)qg * COUT + oBase + l16] = acc[r];
    }
  }
}

// ---------------------------------------------------------------------------
// Launch. d_out = concat(query_xyz[2*8192*3], out[2*8192*64]) per the tuple.
// Workspace layout (~19 MB assumed available in d_ws):
//   [0, 2MB)        idxBuf    int32 (B,S,32)
//   [2MB, 18MB)     inflBuf   f16   (B*S, 16, 32)
//   [18MB, +128KB)  whPacked  f16   (32,4,32,16)  B-fragment order
// ---------------------------------------------------------------------------
extern "C" void kernel_launch(void* const* d_in, const int* in_sizes, int n_in,
                              void* d_out, int out_size, void* d_ws,
                              size_t ws_size, hipStream_t stream) {
  const float* xyz  = (const float*)d_in[0];
  const float* feat = (const float*)d_in[1];
  const float* w    = (const float*)d_in[2];
  const float* kpts = (const float*)d_in[3];
  float* out = (float*)d_out;

  const size_t idxBytes  = (size_t)B_ * S_ * NS * sizeof(int);
  const size_t inflBytes = (size_t)B_ * S_ * KP16 * NS * sizeof(_Float16);
  int*      idxBuf   = (int*)d_ws;
  _Float16* inflBuf  = (_Float16*)((char*)d_ws + idxBytes);
  _Float16* whPacked = (_Float16*)((char*)d_ws + idxBytes + inflBytes);

  // Tuple output element 0: query_xyz passthrough.
  hipMemcpyAsync(d_out, d_in[0], (size_t)B_ * S_ * 3 * sizeof(float),
                 hipMemcpyDeviceToDevice, stream);

  kp_prep_w<<<dim3(256), dim3(256), 0, stream>>>(w, whPacked);
  kp_select<<<dim3((B_ * S_) / QT), dim3(QT), 0, stream>>>(xyz, kpts, idxBuf,
                                                           inflBuf);
  kp_gemm<<<dim3((B_ * S_) / 16), dim3(128), 0, stream>>>(
      feat, idxBuf, inflBuf, whPacked, out + (size_t)B_ * S_ * 3);
}
